// LSTMEncoder_31937376813720
// MI455X (gfx1250) — matrix operand order repaired and checked by reading, not compiled, not run on hardware
//
#include <hip/hip_runtime.h>
#include <hip/hip_bf16.h>

// Sizes from the reference
#define BB 64
#define SS 512
#define FF 128      // raw feature dim
#define DD 512      // dense units
#define UU 1024     // enc units
#define GG 4096     // 4*UU (i,f,g,o gates)
#define KTOT (FF + UU)   // fused K = 1152
#define NKK  (KTOT / 32) // 36 K-chunks of 32

typedef __attribute__((ext_vector_type(16))) __bf16 v16bf;
typedef __attribute__((ext_vector_type(8)))  __bf16 v8bf;
typedef __attribute__((ext_vector_type(8)))  float  v8f;

__device__ __forceinline__ __bf16 f2bf(float f) {
  unsigned u = __builtin_bit_cast(unsigned, f);
  u += 0x7FFFu + ((u >> 16) & 1u);          // round-to-nearest-even
  unsigned short s = (unsigned short)(u >> 16);
  return __builtin_bit_cast(__bf16, s);
}

// ---------- one-time prep kernels ----------

// W_effT[g][f] = sum_d dense_W[f][d] * Wx[d][g]   (stored N-major, bf16)
__global__ void prep_weff(const float* __restrict__ dense_W,
                          const float* __restrict__ Wx,
                          __bf16* __restrict__ WeffT) {
  int idx = blockIdx.x * blockDim.x + threadIdx.x;
  if (idx >= GG * FF) return;
  int g = idx / FF, f = idx % FF;
  const float* wrow = dense_W + (size_t)f * DD;
  float acc = 0.f;
  for (int d = 0; d < DD; ++d) acc += wrow[d] * Wx[(size_t)d * GG + g];
  WeffT[(size_t)g * FF + f] = f2bf(acc);
}

// b_eff[g] = dense_b @ Wx + b
__global__ void prep_beff(const float* __restrict__ dense_b,
                          const float* __restrict__ Wx,
                          const float* __restrict__ b,
                          float* __restrict__ beff) {
  int g = blockIdx.x * blockDim.x + threadIdx.x;
  if (g >= GG) return;
  float acc = b[g];
  for (int d = 0; d < DD; ++d) acc += dense_b[d] * Wx[(size_t)d * GG + g];
  beff[g] = acc;
}

// WhT[g][d] = bf16(Wh[d][g])   (N-major so B-tile K runs are contiguous)
__global__ void prep_wht(const float* __restrict__ Wh, __bf16* __restrict__ WhT) {
  int idx = blockIdx.x * blockDim.x + threadIdx.x;
  if (idx >= GG * UU) return;
  int g = idx / UU, d = idx % UU;
  WhT[(size_t)g * UU + d] = f2bf(Wh[(size_t)d * GG + g]);
}

__global__ void prep_x(const float* __restrict__ x, __bf16* __restrict__ xb, int n) {
  int idx = blockIdx.x * blockDim.x + threadIdx.x;
  if (idx < n) xb[idx] = f2bf(x[idx]);
}

__global__ void prep_state(__bf16* __restrict__ h0, __bf16* __restrict__ h1,
                           float* __restrict__ c) {
  int idx = blockIdx.x * blockDim.x + threadIdx.x;
  if (idx >= BB * UU) return;
  h0[idx] = f2bf(0.f);
  h1[idx] = f2bf(0.f);
  c[idx]  = 0.f;
}

// ---------- per-timestep fused LSTM cell ----------
// Block = 128 threads = 4 waves; wave w owns batch tile (M) w, block owns one
// 16-wide hidden-unit tile (N) for all 4 gates. The 4 waves share the B tiles
// (WhT/WeffT) via a double-buffered LDS stage: global->VGPR issued early,
// ds_store after the WMMAs, one workgroup barrier per K-chunk. This cuts the
// L2-side B traffic 4x vs per-wave fetching.
__global__ __launch_bounds__(128)
void lstm_step(const __bf16* __restrict__ xb,     // [B,S,F] bf16
               const __bf16* __restrict__ WeffT,  // [G,F]   bf16
               const __bf16* __restrict__ WhT,    // [G,U]   bf16
               const float*  __restrict__ beff,   // [G]
               const __bf16* __restrict__ hprev,  // [B,U]   bf16 (read)
               __bf16*       __restrict__ hnext,  // [B,U]   bf16 (write)
               float*        __restrict__ cbuf,   // [B,U]   f32 (rmw, lane-owned)
               float*        __restrict__ out,    // [B,S,U] f32
               int t) {
  __shared__ __bf16 Bs[2][4][512];   // 2 buffers x 4 gates x (16 cols * 32 K) = 8 KB

  const int tid   = threadIdx.x;
  const int lane  = tid & 31;
  const int mtile = tid >> 5;              // wave id == batch tile
  const int utile = (int)blockIdx.x;       // 64 unit tiles of 16
  const int u0    = utile * 16;

  // --- staging roles: 128 threads cover 4 gates x 16 cols x 2 half-chunks ---
  const int sgi   = tid >> 5;              // gate staged by this wave
  const int sj    = tid & 31;
  const int scol  = sj >> 1;               // column within tile (0..15)
  const int spart = sj & 1;                // which 16-half chunk (0/1)
  const size_t ldsoff = (size_t)scol * 32 + (size_t)spart * 16;

  // --- WMMA per-lane addressing ---
  const int rlow = lane & 15;
  const int hi   = (lane >> 4) & 1;
  // A (16-bit, 16x32): lane L and L+16 hold row M=L; 16 contiguous K halves
  // starting at (hi?16:0).
  const int arow = mtile * 16 + rlow;
  const int kc0  = hi ? 16 : 0;
  // B (16-bit, 32x16): lane = (K>=16)*16 + N; 16 contiguous K halves.
  const int bn   = lane & 15;
  const int u    = u0 + bn;

  const __bf16* xrow = xb    + ((size_t)arow * SS + (size_t)t) * FF;
  const __bf16* hrow = hprev + (size_t)arow * UU;

  // stage one B chunk (4KB: 4 gate tiles of 32K x 16N) from global into regs
  auto stage_load = [&](int kk) -> v16bf {
    const int kbase = kk * 32;
    if (kbase < FF)
      return *(const v16bf*)(WeffT + (size_t)(sgi * UU + u0 + scol) * FF +
                             kbase + spart * 16);
    const int kh = kbase - FF;
    return *(const v16bf*)(WhT + (size_t)(sgi * UU + u0 + scol) * UU +
                           kh + spart * 16);
  };

  v8f acc0 = {}, acc1 = {}, acc2 = {}, acc3 = {};

  // prologue: fill buffer 0
  {
    v16bf r = stage_load(0);
    *(v16bf*)&Bs[0][sgi][ldsoff] = r;
  }
  __syncthreads();

  for (int kk = 0; kk < NKK; ++kk) {
    const int cur = kk & 1;
    const int kbase = kk * 32;

    // issue next chunk's global loads early (latency hidden behind WMMAs)
    v16bf nxt;
    if (kk + 1 < NKK) nxt = stage_load(kk + 1);

    // A tile for this wave
    v8bf a_lo, a_hi;
    if (kbase < FF) {
      a_lo = *(const v8bf*)(xrow + kbase + kc0);
      a_hi = *(const v8bf*)(xrow + kbase + kc0 + 8);
    } else {
      const int kh = kbase - FF;
      a_lo = *(const v8bf*)(hrow + kh + kc0);
      a_hi = *(const v8bf*)(hrow + kh + kc0 + 8);
    }
    v16bf a = __builtin_shufflevector(a_lo, a_hi,
              0,1,2,3,4,5,6,7,8,9,10,11,12,13,14,15);

    // B tiles from LDS (broadcast across the 4 waves)
    const size_t boff = (size_t)bn * 32 + (size_t)hi * 16;
    v16bf b0 = *(const v16bf*)&Bs[cur][0][boff];
    v16bf b1 = *(const v16bf*)&Bs[cur][1][boff];
    v16bf b2 = *(const v16bf*)&Bs[cur][2][boff];
    v16bf b3 = *(const v16bf*)&Bs[cur][3][boff];

    acc0 = __builtin_amdgcn_wmma_f32_16x16x32_bf16(false, a, false, b0, (short)0, acc0, false, false);
    acc1 = __builtin_amdgcn_wmma_f32_16x16x32_bf16(false, a, false, b1, (short)0, acc1, false, false);
    acc2 = __builtin_amdgcn_wmma_f32_16x16x32_bf16(false, a, false, b2, (short)0, acc2, false, false);
    acc3 = __builtin_amdgcn_wmma_f32_16x16x32_bf16(false, a, false, b3, (short)0, acc3, false, false);

    // park next chunk in the other LDS buffer, then one barrier per K-chunk
    if (kk + 1 < NKK) {
      *(v16bf*)&Bs[cur ^ 1][sgi][ldsoff] = nxt;
    }
    __syncthreads();
  }

  // Pointwise LSTM cell. C/D tile layout: lane = N (dup over 16), VGPR r holds
  // M = r + hi*8.
  const float bi = beff[u];
  const float bfv = beff[UU + u];
  const float bg = beff[2 * UU + u];
  const float bo = beff[3 * UU + u];

#pragma unroll
  for (int r = 0; r < 8; ++r) {
    const int brow = mtile * 16 + r + hi * 8;
    const float zi = acc0[r] + bi;
    const float zf = acc1[r] + bfv;
    const float zg = acc2[r] + bg;
    const float zo = acc3[r] + bo;
    const float ig = 1.f / (1.f + __expf(-zi));
    const float fg = 1.f / (1.f + __expf(-zf));
    const float gg = tanhf(zg);
    const float og = 1.f / (1.f + __expf(-zo));
    const size_t su = (size_t)brow * UU + u;
    const float cn = fg * cbuf[su] + ig * gg;
    const float hn = og * tanhf(cn);
    cbuf[su]  = cn;
    hnext[su] = f2bf(hn);
    out[((size_t)brow * SS + (size_t)t) * UU + u] = hn;
  }
}

// ---------- host launch ----------
extern "C" void kernel_launch(void* const* d_in, const int* in_sizes, int n_in,
                              void* d_out, int out_size, void* d_ws, size_t ws_size,
                              hipStream_t stream) {
  (void)in_sizes; (void)n_in; (void)out_size; (void)ws_size;
  const float* x       = (const float*)d_in[0];
  const float* dense_W = (const float*)d_in[1];
  const float* dense_b = (const float*)d_in[2];
  const float* Wx      = (const float*)d_in[3];
  const float* Wh      = (const float*)d_in[4];
  const float* b       = (const float*)d_in[5];
  float* out = (float*)d_out;

  // Workspace layout (all chunk sizes are multiples of 256 bytes)
  char* p = (char*)d_ws;
  __bf16* WhT   = (__bf16*)p; p += (size_t)GG * UU * sizeof(__bf16);   // 8 MB
  __bf16* WeffT = (__bf16*)p; p += (size_t)GG * FF * sizeof(__bf16);   // 1 MB
  float*  beff  = (float*)p;  p += (size_t)GG * sizeof(float);         // 16 KB
  __bf16* xbuf  = (__bf16*)p; p += (size_t)BB * SS * FF * sizeof(__bf16); // 8 MB
  __bf16* h0    = (__bf16*)p; p += (size_t)BB * UU * sizeof(__bf16);   // 128 KB
  __bf16* h1    = (__bf16*)p; p += (size_t)BB * UU * sizeof(__bf16);   // 128 KB
  float*  cbuf  = (float*)p;  p += (size_t)BB * UU * sizeof(float);    // 256 KB

  const int TPB = 256;
  prep_weff <<<(GG * FF + TPB - 1) / TPB, TPB, 0, stream>>>(dense_W, Wx, WeffT);
  prep_beff <<<(GG + TPB - 1) / TPB,      TPB, 0, stream>>>(dense_b, Wx, b, beff);
  prep_wht  <<<(GG * UU + TPB - 1) / TPB, TPB, 0, stream>>>(Wh, WhT);
  prep_x    <<<(BB * SS * FF + TPB - 1) / TPB, TPB, 0, stream>>>(x, xbuf, BB * SS * FF);
  prep_state<<<(BB * UU + TPB - 1) / TPB, TPB, 0, stream>>>(h0, h1, cbuf);

  // Sequential scan: 64 blocks x 128 threads per step; h double-buffered.
  for (int t = 0; t < SS; ++t) {
    __bf16* hp = (t & 1) ? h1 : h0;
    __bf16* hn = (t & 1) ? h0 : h1;
    lstm_step<<<64, 128, 0, stream>>>(xbuf, WeffT, WhT, beff, hp, hn, cbuf, out, t);
  }
}
